// STMM_Cell_8675833938741
// MI455X (gfx1250) — matrix operand order repaired and checked by reading, not compiled, not run on hardware
//
#include <hip/hip_runtime.h>
#include <hip/hip_bf16.h>
#include <math.h>

// ---------------------------------------------------------------------------
// STMM cell for MI455X (gfx1250, wave32, WMMA).
// GEMM path: bf16 operands, f32 accumulate via v_wmma_f32_16x16x32_bf16.
// Each wave computes a 16x32 output tile: one A fragment feeds two WMMAs into
// two independent accumulators (two dependency chains, fewer loads per FLOP).
// Activations are staged transposed ([pixel][channel], bf16) so every WMMA
// fragment is filled with contiguous 16B loads per lane (ISA 7.12.2 layouts).
// Workspace requirement: ~340 MB (aligned/zt/rt f32 + bf16 staging).
// ---------------------------------------------------------------------------

typedef __bf16 bf16_t;
typedef __attribute__((ext_vector_type(8)))  bf16_t v8bf;
typedef __attribute__((ext_vector_type(16))) bf16_t v16bf;
typedef __attribute__((ext_vector_type(8)))  float  v8f;

#define CIN   2048
#define COUT  1024
#define BDIM  4
#define HDIM  64
#define WDIM  64
#define NPIX  (HDIM * WDIM)          // 4096 pixels per batch image

// float -> bf16, round-to-nearest-even
__device__ __forceinline__ bf16_t f2bf(float f) {
    union { float f; unsigned u; } x; x.f = f;
    unsigned r = x.u + 0x7FFFu + ((x.u >> 16) & 1u);
    unsigned short h = (unsigned short)(r >> 16);
    return __builtin_bit_cast(bf16_t, h);
}

// -------------------- match_trans helpers --------------------

// nc[b,n] = sum_c x[b,c,n]   (coalesced: consecutive threads = consecutive n)
__global__ __launch_bounds__(256)
void channel_sum_kernel(const float* __restrict__ x, float* __restrict__ nc, int C) {
    const int b = blockIdx.y;
    const int n = blockIdx.x * blockDim.x + threadIdx.x;
    const float* p = x + (size_t)b * C * NPIX + n;
    float s = 0.0f;
    for (int c = 0; c < C; ++c) s += p[(size_t)c * NPIX];
    nc[b * NPIX + n] = s;
}

// ss[b,y,x] = 5x5 zero-padded window sum of nc
__global__ __launch_bounds__(256)
void local_sum5_kernel(const float* __restrict__ nc, float* __restrict__ ss) {
    const int b = blockIdx.y;
    const int n = blockIdx.x * blockDim.x + threadIdx.x;
    const int y = n / WDIM, x = n % WDIM;
    float s = 0.0f;
    for (int dy = -2; dy <= 2; ++dy) {
        int yy = y + dy; if (yy < 0 || yy >= HDIM) continue;
        for (int dx = -2; dx <= 2; ++dx) {
            int xx = x + dx; if (xx < 0 || xx >= WDIM) continue;
            s += nc[b * NPIX + yy * WDIM + xx];
        }
    }
    ss[b * NPIX + n] = s;
}

// aligned[b,c,y,x] = sum_{dy,dx} mask(b,y,x,dy,dx) * prev_M[b,c,y+dy,x+dx]
// mask = curr_nc * prev_nc(y+dy,x+dx) / (curr_nc * C_IN * window_sum)
// Block = one (b,y) row: 64 x-threads * 4 channel groups; masks cached in LDS.
__global__ __launch_bounds__(256)
void align_kernel(const float* __restrict__ curr_nc,
                  const float* __restrict__ prev_nc,
                  const float* __restrict__ ssum,
                  const float* __restrict__ prev_M,
                  const int*   __restrict__ body,
                  float* __restrict__ aligned)
{
    __shared__ float smask[25][64];
    __shared__ int   soff [25][64];
    const int tx = threadIdx.x & 63;
    const int tz = threadIdx.x >> 6;          // 0..3
    const int b  = blockIdx.x / HDIM;
    const int y  = blockIdx.x % HDIM;
    const int n  = y * WDIM + tx;

    if (*body == 0) {                         // aligned = prev_M
        for (int c = tz; c < COUT; c += 4) {
            size_t idx = (size_t)b * COUT * NPIX + (size_t)c * NPIX + n;
            aligned[idx] = prev_M[idx];
        }
        return;
    }

    if (threadIdx.x < 64) {
        float cn    = curr_nc[b * NPIX + n];
        float ss    = ssum[b * NPIX + n];
        float denom = cn * (2048.0f * ss);
        float inv   = (denom != 0.0f) ? (cn / denom) : 0.0f;
        int t = 0;
        for (int dy = -2; dy <= 2; ++dy)
            for (int dx = -2; dx <= 2; ++dx, ++t) {
                int yy = y + dy, xx = tx + dx;
                bool ok = (yy >= 0) && (yy < HDIM) && (xx >= 0) && (xx < WDIM);
                int off = ok ? (yy * WDIM + xx) : 0;
                smask[t][tx] = ok ? (prev_nc[b * NPIX + off] * inv) : 0.0f;
                soff [t][tx] = off;
            }
    }
    __syncthreads();

    const float* pM = prev_M + (size_t)b * COUT * NPIX;
    for (int c = tz; c < COUT; c += 4) {
        const float* pc = pM + (size_t)c * NPIX;
        float acc = 0.0f;
        #pragma unroll
        for (int t = 0; t < 25; ++t)
            acc += smask[t][tx] * pc[soff[t][tx]];
        aligned[(size_t)b * COUT * NPIX + (size_t)c * NPIX + n] = acc;
    }
}

// -------------------- bf16 staging --------------------

__global__ __launch_bounds__(256)
void f32_to_bf16_kernel(const float* __restrict__ in, bf16_t* __restrict__ out, int n) {
    int i = blockIdx.x * blockDim.x + threadIdx.x;
    if (i < n) out[i] = f2bf(in[i]);
}

// in: [B][C][N] f32 (optional elementwise mul) -> out: [B][N][C] bf16 (LDS tiled)
__global__ __launch_bounds__(256)
void transpose_to_bf16_kernel(const float* __restrict__ in,
                              const float* __restrict__ mul,  // may be null
                              bf16_t* __restrict__ out, int C)
{
    __shared__ float tile[32][33];
    const int b  = blockIdx.z;
    const int n0 = blockIdx.x * 32;
    const int c0 = blockIdx.y * 32;
    const int tx = threadIdx.x, ty = threadIdx.y;   // 32 x 8
    const float* ip = in + (size_t)b * C * NPIX;
    const float* mp = mul ? mul + (size_t)b * C * NPIX : nullptr;
    #pragma unroll
    for (int i = 0; i < 32; i += 8) {
        size_t idx = (size_t)(c0 + ty + i) * NPIX + (n0 + tx);
        float v = ip[idx];
        if (mp) v *= mp[idx];
        tile[ty + i][tx] = v;
    }
    __syncthreads();
    bf16_t* op = out + (size_t)b * C * NPIX;
    #pragma unroll
    for (int i = 0; i < 32; i += 8)
        op[(size_t)(n0 + ty + i) * C + (c0 + tx)] = f2bf(tile[tx][ty + i]);
}

// -------------------- WMMA GEMM --------------------

__device__ __forceinline__ v16bf load_frag_a(const bf16_t* p) {
    v8bf lo = *(const v8bf*)(p);
    v8bf hi = *(const v8bf*)(p + 16);
    return __builtin_shufflevector(lo, hi, 0,1,2,3,4,5,6,7,8,9,10,11,12,13,14,15);
}
__device__ __forceinline__ v16bf load_frag_b(const bf16_t* p) {
    v8bf lo = *(const v8bf*)(p);
    v8bf hi = *(const v8bf*)(p + 16);
    return __builtin_shufflevector(lo, hi, 0,1,2,3,4,5,6,7,8,9,10,11,12,13,14,15);
}

// One K-sweep: {acc0, acc1} += A * {B0, B1}.  A fragment reused across the two
// WMMAs; the two accumulators form independent dependency chains.
// A: weight rows [o][k] — lane<16 holds K0..7 & K16..23, lane>=16 K8..15 & K24..31.
// B: pixel rows  [n][k] — lane<16 holds K0..7 & K16..23 of col n=lane, lane>=16 K8..15 & K24..31.
__device__ __forceinline__ void wmma_k_loop(const bf16_t* __restrict__ aptr,
                                            const bf16_t* __restrict__ bptr0,
                                            const bf16_t* __restrict__ bptr1,
                                            int K, v8f& acc0, v8f& acc1)
{
    for (int k = 0; k < K; k += 32) {
        __builtin_prefetch(bptr0 + k + 128, 0, 0);
        __builtin_prefetch(bptr1 + k + 128, 0, 0);
        v16bf A  = load_frag_a(aptr + k);
        v16bf B0 = load_frag_b(bptr0 + k);
        v16bf B1 = load_frag_b(bptr1 + k);
        acc0 = __builtin_amdgcn_wmma_f32_16x16x32_bf16(false, A, false, B0,
                                                       (short)0, acc0, false, false);
        acc1 = __builtin_amdgcn_wmma_f32_16x16x32_bf16(false, A, false, B1,
                                                       (short)0, acc1, false, false);
    }
}

// out[b,o,n] = epilogue( sum_k Wa[o,k]*Xa[b,n,k] + sum_k Wu[o,k]*Xu[b,n,k] + b0[o]+b1[o] )
// EPI=0: relu-store.  EPI=1: M=relu(..); out = (1-zt)*aligned + zt*M.
// Block: 256 threads = 8 waves (4 along O x 2 along N); block tile 64x64,
// each wave owns a 16x32 tile (two 16x16 accumulators).
template <int EPI>
__global__ __launch_bounds__(256)
void gemm_dual_wmma(const bf16_t* __restrict__ Wa, const bf16_t* __restrict__ Xa, int K1,
                    const bf16_t* __restrict__ Wu, const bf16_t* __restrict__ Xu, int K2,
                    const float* __restrict__ bias0, const float* __restrict__ bias1,
                    const float* __restrict__ zt, const float* __restrict__ aligned,
                    float* __restrict__ out)
{
    const int b    = blockIdx.z;
    const int lane = threadIdx.x & 31;
    const int wave = threadIdx.x >> 5;
    const int wm   = wave & 3, wn = wave >> 2;
    const int o0   = blockIdx.x * 64 + wm * 16;
    const int n0   = blockIdx.y * 64 + wn * 32;     // wave tile: 16 (O) x 32 (N)

    const int row    = lane & 15;
    const int ka_off = (lane < 16) ? 0 : 8;
    const int kb_off = (lane < 16) ? 0 : 8;

    v8f acc0 = {}, acc1 = {};
    {
        const bf16_t* a  = Wa + (size_t)(o0 + row) * K1 + ka_off;
        const bf16_t* b0 = Xa + ((size_t)b * NPIX + (n0 + row)) * K1 + kb_off;
        const bf16_t* b1 = b0 + (size_t)16 * K1;
        wmma_k_loop(a, b0, b1, K1, acc0, acc1);
    }
    {
        const bf16_t* a  = Wu + (size_t)(o0 + row) * K2 + ka_off;
        const bf16_t* b0 = Xu + ((size_t)b * NPIX + (n0 + row)) * K2 + kb_off;
        const bf16_t* b1 = b0 + (size_t)16 * K2;
        wmma_k_loop(a, b0, b1, K2, acc0, acc1);
    }

    const int mbase = (lane >> 4) << 3;   // lanes 16..31 hold M = r+8
    const int nc0   = n0 + (lane & 15);
    #pragma unroll
    for (int r = 0; r < 8; ++r) {
        const int o = o0 + r + mbase;
        const float bsum = bias0[o] + bias1[o];
        const size_t rowb = (size_t)b * COUT * NPIX + (size_t)o * NPIX;
        #pragma unroll
        for (int t = 0; t < 2; ++t) {
            float v = (t == 0 ? acc0[r] : acc1[r]) + bsum;
            v = v > 0.0f ? v : 0.0f;               // relu
            size_t idx = rowb + nc0 + t * 16;
            if constexpr (EPI == 0) {
                out[idx] = v;
            } else {
                float z  = zt[idx];
                float al = aligned[idx];
                out[idx] = (1.0f - z) * al + z * v;
            }
        }
    }
}

// -------------------- bn_star (batch-axis stats, B=4, ddof=1) --------------------

__global__ __launch_bounds__(256)
void bn_star_kernel(float* __restrict__ x) {
    size_t t = (size_t)blockIdx.x * blockDim.x + threadIdx.x;
    const size_t s = (size_t)COUT * NPIX;
    if (t >= s) return;
    float x0 = x[t], x1 = x[t + s], x2 = x[t + 2 * s], x3 = x[t + 3 * s];
    float mu = 0.25f * (x0 + x1 + x2 + x3);
    float d0 = x0 - mu, d1 = x1 - mu, d2 = x2 - mu, d3 = x3 - mu;
    float sd = sqrtf((d0 * d0 + d1 * d1 + d2 * d2 + d3 * d3) * (1.0f / 3.0f));
    float lim = mu + 3.0f * sd;
    x[t]         = (x0 > lim) ? 1.0f : x0;
    x[t + s]     = (x1 > lim) ? 1.0f : x1;
    x[t + 2 * s] = (x2 > lim) ? 1.0f : x2;
    x[t + 3 * s] = (x3 > lim) ? 1.0f : x3;
}

// -------------------- host launcher --------------------

static inline size_t align256(size_t x) { return (x + 255) & ~(size_t)255; }

extern "C" void kernel_launch(void* const* d_in, const int* in_sizes, int n_in,
                              void* d_out, int out_size, void* d_ws, size_t ws_size,
                              hipStream_t stream)
{
    (void)in_sizes; (void)n_in; (void)out_size; (void)ws_size;
    const float* curr_F = (const float*)d_in[0];
    const float* prev_F = (const float*)d_in[1];
    const float* prev_M = (const float*)d_in[2];
    const float* Wz_w   = (const float*)d_in[3];
    const float* Wz_b   = (const float*)d_in[4];
    const float* Wr_w   = (const float*)d_in[5];
    const float* Wr_b   = (const float*)d_in[6];
    const float* W_w    = (const float*)d_in[7];
    const float* W_b    = (const float*)d_in[8];
    const float* Uz_w   = (const float*)d_in[9];
    const float* Uz_b   = (const float*)d_in[10];
    const float* Ur_w   = (const float*)d_in[11];
    const float* Ur_b   = (const float*)d_in[12];
    const float* U_w    = (const float*)d_in[13];
    const float* U_b    = (const float*)d_in[14];
    const int*   body   = (const int*)d_in[15];
    float* out = (float*)d_out;

    // workspace carve
    char* ws = (char*)d_ws;
    size_t off = 0;
    auto carve = [&](size_t bytes) { char* p = ws + off; off += align256(bytes); return p; };
    float*  curr_nc = (float*)carve((size_t)BDIM * NPIX * 4);
    float*  prev_nc = (float*)carve((size_t)BDIM * NPIX * 4);
    float*  ssum    = (float*)carve((size_t)BDIM * NPIX * 4);
    float*  alignedF= (float*)carve((size_t)BDIM * COUT * NPIX * 4);   // 64 MB
    float*  tmpZ    = (float*)carve((size_t)BDIM * COUT * NPIX * 4);   // 64 MB
    float*  tmpR    = (float*)carve((size_t)BDIM * COUT * NPIX * 4);   // 64 MB
    bf16_t* currT   = (bf16_t*)carve((size_t)BDIM * NPIX * CIN  * 2);  // 64 MB
    bf16_t* alT     = (bf16_t*)carve((size_t)BDIM * NPIX * COUT * 2);  // 32 MB
    bf16_t* arT     = (bf16_t*)carve((size_t)BDIM * NPIX * COUT * 2);  // 32 MB
    bf16_t* Wzbf    = (bf16_t*)carve((size_t)COUT * CIN  * 2);
    bf16_t* Wrbf    = (bf16_t*)carve((size_t)COUT * CIN  * 2);
    bf16_t* Wbf     = (bf16_t*)carve((size_t)COUT * CIN  * 2);
    bf16_t* Uzbf    = (bf16_t*)carve((size_t)COUT * COUT * 2);
    bf16_t* Urbf    = (bf16_t*)carve((size_t)COUT * COUT * 2);
    bf16_t* Ubf     = (bf16_t*)carve((size_t)COUT * COUT * 2);

    // 1) channel sums + 5x5 window sum + alignment masks
    dim3 gpix(NPIX / 256, BDIM);
    channel_sum_kernel<<<gpix, 256, 0, stream>>>(curr_F, curr_nc, CIN);
    channel_sum_kernel<<<gpix, 256, 0, stream>>>(prev_F, prev_nc, CIN);
    local_sum5_kernel<<<gpix, 256, 0, stream>>>(prev_nc, ssum);
    align_kernel<<<BDIM * HDIM, 256, 0, stream>>>(curr_nc, prev_nc, ssum, prev_M, body, alignedF);

    // 2) stage weights + activations as bf16
    const int nw1 = COUT * CIN, nw2 = COUT * COUT;
    f32_to_bf16_kernel<<<nw1 / 256, 256, 0, stream>>>(Wz_w, Wzbf, nw1);
    f32_to_bf16_kernel<<<nw1 / 256, 256, 0, stream>>>(Wr_w, Wrbf, nw1);
    f32_to_bf16_kernel<<<nw1 / 256, 256, 0, stream>>>(W_w,  Wbf,  nw1);
    f32_to_bf16_kernel<<<nw2 / 256, 256, 0, stream>>>(Uz_w, Uzbf, nw2);
    f32_to_bf16_kernel<<<nw2 / 256, 256, 0, stream>>>(Ur_w, Urbf, nw2);
    f32_to_bf16_kernel<<<nw2 / 256, 256, 0, stream>>>(U_w,  Ubf,  nw2);
    transpose_to_bf16_kernel<<<dim3(NPIX / 32, CIN  / 32, BDIM), dim3(32, 8), 0, stream>>>(
        curr_F, nullptr, currT, CIN);
    transpose_to_bf16_kernel<<<dim3(NPIX / 32, COUT / 32, BDIM), dim3(32, 8), 0, stream>>>(
        alignedF, nullptr, alT, COUT);

    // 3) z / r pre-activations (fused dual GEMM + bias + relu)
    dim3 ggrid(COUT / 64, NPIX / 64, BDIM);
    gemm_dual_wmma<0><<<ggrid, 256, 0, stream>>>(Wzbf, currT, CIN, Uzbf, alT, COUT,
                                                 Wz_b, Uz_b, nullptr, nullptr, tmpZ);
    gemm_dual_wmma<0><<<ggrid, 256, 0, stream>>>(Wrbf, currT, CIN, Urbf, alT, COUT,
                                                 Wr_b, Ur_b, nullptr, nullptr, tmpR);

    // 4) bn_star in place -> zt, rt
    const int nbn = (COUT * NPIX + 255) / 256;
    bn_star_kernel<<<nbn, 256, 0, stream>>>(tmpZ);
    bn_star_kernel<<<nbn, 256, 0, stream>>>(tmpR);

    // 5) aligned * rt -> bf16 [n][c]
    transpose_to_bf16_kernel<<<dim3(NPIX / 32, COUT / 32, BDIM), dim3(32, 8), 0, stream>>>(
        alignedF, tmpR, arT, COUT);

    // 6) M GEMM + fused GRU combine -> curr_M
    gemm_dual_wmma<1><<<ggrid, 256, 0, stream>>>(Wbf, currT, CIN, Ubf, arT, COUT,
                                                 W_b, U_b, tmpZ, alignedF, out);
}
